// Attention_16793322128075
// MI455X (gfx1250) — matrix operand order repaired and checked by reading, not compiled, not run on hardware
//
#include <hip/hip_runtime.h>
#include <hip/hip_bf16.h>

// ---------------------------------------------------------------------------
// Gemma-style sliding-window GQA attention prefill for gfx1250 (MI455X).
// All matmuls run through V_WMMA_F32_16X16X32_BF16 (f32 accumulation).
// Straight-copy LDS stagings use GLOBAL_LOAD_ASYNC_TO_LDS_B128 (ASYNCcnt).
// ---------------------------------------------------------------------------

typedef __attribute__((ext_vector_type(16))) __bf16 v16bf;
typedef __attribute__((ext_vector_type(8)))  float  v8f;

// ---- problem constants (match reference setup_inputs) ----
#define TT   4096
#define DD   2048
#define NH   16      // query heads
#define KVH  8       // kv heads
#define HH   128     // head dim
#define WIN  1024
#define ATTN_SCALE 0.08838834764831845f
#define LN10000    9.210340371976184f   // ln(10000)

__device__ __forceinline__ v8f wmma_bf16(v16bf a, v16bf b, v8f c) {
  // D = A(16x32) * B(32x16) + C, f32 accum
  return __builtin_amdgcn_wmma_f32_16x16x32_bf16(
      /*neg_a=*/false, a, /*neg_b=*/false, b,
      /*c_mod=*/(short)0, c, /*reuse_a=*/false, /*reuse_b=*/false);
}

// Async copy of 16 bytes global -> LDS (GV mode, per-lane addresses).
// The generic pointer to a __shared__ object is SHARED_BASE | lds_offset, so
// truncation to 32 bits yields the wave-relative LDS byte address the async
// unit expects (CDNA5 ISA).
__device__ __forceinline__ void async_ld_b128(void* lds, const void* g) {
  unsigned      loff = (unsigned)(unsigned long long)(uintptr_t)lds;
  unsigned long long ga = (unsigned long long)(uintptr_t)g;
  asm volatile("global_load_async_to_lds_b128 %0, %1, off"
               :: "v"(loff), "v"(ga) : "memory");
}
__device__ __forceinline__ void wait_async0() {
  asm volatile("s_wait_asynccnt 0" ::: "memory");
}

// A fragment (16x32 bf16, MxK). ISA layout: lanes 0-15 hold row m=lane with
// K = {0..7, 16..23}; lanes 16-31 hold row m=lane-16 with K = {8..15, 24..31}.
__device__ __forceinline__ v16bf load_frag_a(const __bf16* base, int stride, int lane) {
  const int m    = lane & 15;
  const int koff = (lane < 16) ? 0 : 8;
  const __bf16* p = base + m * stride + koff;
  v16bf f;
#pragma unroll
  for (int e = 0; e < 8; ++e) f[e] = p[e];
#pragma unroll
  for (int e = 0; e < 8; ++e) f[8 + e] = p[16 + e];
  return f;
}

// B fragment (32x16 bf16, KxN), stored column-major in LDS (K contiguous per
// column). lane holds column n=lane%16; lanes 0-15: K={0..7,16..23}, lanes 16-31:
// K={8..15,24..31}.
__device__ __forceinline__ v16bf load_frag_b(const __bf16* base, int stride, int lane) {
  const int n    = lane & 15;
  const int koff = (lane < 16) ? 0 : 8;
  const __bf16* p = base + n * stride + koff;
  v16bf f;
#pragma unroll
  for (int e = 0; e < 8; ++e) f[e] = p[e];
#pragma unroll
  for (int e = 0; e < 8; ++e) f[8 + e] = p[16 + e];
  return f;
}

// ---------------------------------------------------------------------------
// f32 -> bf16 conversion
// ---------------------------------------------------------------------------
__global__ void cvt_f32_bf16(const float* __restrict__ in,
                             __bf16* __restrict__ out, long long n) {
  long long i = (long long)blockIdx.x * blockDim.x + threadIdx.x;
  if (i < n) out[i] = (__bf16)in[i];
}

// ---------------------------------------------------------------------------
// Generic 64(M) x 128(N) tile GEMM:  C = A[M x K] * B[K x N]
// EPI: 0 = store f32, 1 = store bf16, 2 = RMSNorm(H=128) + RoPE then bf16
// Block = 256 threads (8 wave32); wave w owns mi = w/2 (16 rows) and
// hbase = (w&1)*64 (4 x 16 cols) -> 4 v8f accumulators.
// ---------------------------------------------------------------------------
template <int EPI>
__global__ __launch_bounds__(256)
void gemm64x128(const __bf16* __restrict__ A, int lda,
                const __bf16* __restrict__ B, long long bcolStride, int ldb,
                int Kdim,
                void* __restrict__ Out, long long ocolStride, long long ldo,
                const float* __restrict__ normScale) {
  __shared__ __bf16 As[64 * 40];   // row-major 64 x 32, stride 40 halves (80B, 16B-aligned rows)
  __shared__ __bf16 Bs[128 * 40];  // column-major: col c, k -> Bs[c*40 + k]

  const int tid  = threadIdx.x;
  const int lane = tid & 31;
  const int wave = tid >> 5;
  const int row0 = blockIdx.x * 64;
  const __bf16* Bblk = B + (long long)blockIdx.y * bcolStride;

  const int mi    = wave >> 1;
  const int hbase = (wave & 1) * 64;

  v8f acc[4] = {};

  for (int k0 = 0; k0 < Kdim; k0 += 32) {
    // ---- stage A slab async: 64 rows x 32 halves (16B per thread) ----
    {
      const int r = tid >> 2;
      const int c = (tid & 3) * 8;
      async_ld_b128(As + r * 40 + c,
                    A + (long long)(row0 + r) * lda + k0 + c);
    }
    // ---- stage B slab transposed (sync): 32 k-rows x 128 cols ----
#pragma unroll
    for (int e = 0; e < 16; ++e) {
      const int idx = e * 256 + tid;   // 0..4095
      const int c   = idx & 127;
      const int kk  = idx >> 7;        // 0..31
      Bs[c * 40 + kk] = Bblk[(long long)(k0 + kk) * ldb + c];
    }
    wait_async0();
    __syncthreads();

    const v16bf afrag = load_frag_a(As + mi * 16 * 40, 40, lane);
#pragma unroll
    for (int h = 0; h < 4; ++h) {
      const v16bf bfrag = load_frag_b(Bs + (hbase + h * 16) * 40, 40, lane);
      acc[h] = wmma_bf16(afrag, bfrag, acc[h]);
    }
    __syncthreads();
  }

  if constexpr (EPI != 2) {
#pragma unroll
    for (int h = 0; h < 4; ++h) {
      const int cc = hbase + h * 16 + (lane & 15);
#pragma unroll
      for (int j = 0; j < 8; ++j) {
        const int r = mi * 16 + j + ((lane & 16) ? 8 : 0);
        const long long off =
            (long long)(row0 + r) * ldo + (long long)blockIdx.y * ocolStride + cc;
        if constexpr (EPI == 0) ((float*)Out)[off]   = acc[h][j];
        else                    ((__bf16*)Out)[off]  = (__bf16)acc[h][j];
      }
    }
  } else {
    // ---- fused RMSNorm (over H=128) + RoPE epilogue ----
    __shared__ float Ct[64 * 132];
#pragma unroll
    for (int h = 0; h < 4; ++h) {
      const int cc = hbase + h * 16 + (lane & 15);
#pragma unroll
      for (int j = 0; j < 8; ++j) {
        const int r = mi * 16 + j + ((lane & 16) ? 8 : 0);
        Ct[r * 132 + cc] = acc[h][j];
      }
    }
    __syncthreads();
    if (tid < 64) {
      const float* rowp = Ct + tid * 132;
      float ss = 0.f;
#pragma unroll 8
      for (int c = 0; c < 128; ++c) ss += rowp[c] * rowp[c];
      const float inv = rsqrtf(ss * (1.0f / 128.0f) + 1e-6f);
      const int pos = row0 + tid;
      __bf16* O = (__bf16*)Out + (long long)blockIdx.y * ocolStride +
                  (long long)pos * ldo;
      for (int h = 0; h < 64; ++h) {
        const float x1 = rowp[h] * inv * normScale[h];
        const float x2 = rowp[h + 64] * inv * normScale[h + 64];
        const float invts = __expf(-(float)h * (LN10000 / 64.0f));
        const float ang = (float)pos * invts;
        const float s = __sinf(ang), c = __cosf(ang);
        O[h]      = (__bf16)(x1 * c - x2 * s);
        O[h + 64] = (__bf16)(x2 * c + x1 * s);
      }
    }
  }
}

// ---------------------------------------------------------------------------
// Flash attention with causal + sliding-window mask.
// grid = (T/64, N). Block 256 (8 wave32).
// ---------------------------------------------------------------------------
__global__ __launch_bounds__(256)
void attn_kernel(const __bf16* __restrict__ qb, const __bf16* __restrict__ kb,
                 const __bf16* __restrict__ vb, __bf16* __restrict__ encb) {
  const int it = blockIdx.x;
  const int n  = blockIdx.y;
  const int kh = n >> 1;  // G = N/K = 2
  const int i0 = it * 64;
  const int tid = threadIdx.x, lane = tid & 31, wave = tid >> 5;

  // stride 136 halves = 272B = 17*16B: async B128-aligned rows, 4-dword bank rotation
  __shared__ __bf16 Qs[64 * 136];   // [q row][h]
  __shared__ __bf16 Ks[64 * 136];   // [k row][h]  (B for Q*K^T)
  __shared__ __bf16 Vt[128 * 68];   // [h][s]      (column-major B for P*V)
  __shared__ float  Ss[64 * 66];    // logits tile
  __shared__ __bf16 Ps[64 * 68];    // probs tile (A for P*V)
  __shared__ float  mrow[64], lrow[64], arow[64];

  // ---- load Q tile async: 64 rows x 256B = 1024 x 16B chunks ----
#pragma unroll
  for (int e = 0; e < 4; ++e) {
    const int chunk = e * 256 + tid;       // 0..1023
    const int r   = chunk >> 4;            // 16 chunks per row
    const int c8  = (chunk & 15) * 8;      // halves offset
    async_ld_b128(Qs + r * 136 + c8,
                  qb + (long long)(i0 + r) * (NH * HH) + n * HH + c8);
  }
  if (tid < 64) { mrow[tid] = -1e30f; lrow[tid] = 0.f; }

  const int mi    = wave >> 1;
  const int hbase = (wave & 1) * 64;
  const int nib   = (wave & 1) * 2;
  v8f Oacc[4] = {};
  wait_async0();
  __syncthreads();

  int jlo = i0 - (WIN - 1); if (jlo < 0) jlo = 0;
  const int jt0 = jlo >> 6;

  for (int jt = jt0; jt <= it; ++jt) {
    const int s0 = jt * 64;
    // ---- K tile async (straight copy), V tile sync (transposed) ----
#pragma unroll
    for (int e = 0; e < 4; ++e) {
      const int chunk = e * 256 + tid;
      const int r  = chunk >> 4;
      const int c8 = (chunk & 15) * 8;
      async_ld_b128(Ks + r * 136 + c8,
                    kb + (long long)(s0 + r) * (KVH * HH) + kh * HH + c8);
    }
#pragma unroll
    for (int e = 0; e < 32; ++e) {
      const int idx = e * 256 + tid;
      const int r = idx >> 7, h = idx & 127;
      Vt[h * 68 + r] = vb[(long long)(s0 + r) * (KVH * HH) + kh * HH + h];
    }
    wait_async0();
    __syncthreads();

    // ---- S = Q * K^T : wave handles tiles (mi, nib..nib+1) of 64x64 ----
    v8f Sacc[2] = {};
#pragma unroll
    for (int ks = 0; ks < 4; ++ks) {
      const v16bf a = load_frag_a(Qs + mi * 16 * 136 + ks * 32, 136, lane);
#pragma unroll
      for (int ni = 0; ni < 2; ++ni) {
        const v16bf b =
            load_frag_b(Ks + (nib + ni) * 16 * 136 + ks * 32, 136, lane);
        Sacc[ni] = wmma_bf16(a, b, Sacc[ni]);
      }
    }
    // scale + causal/sliding-window mask, spill to LDS
#pragma unroll
    for (int ni = 0; ni < 2; ++ni) {
      const int cl = (nib + ni) * 16 + (lane & 15);
      const int cg = s0 + cl;
#pragma unroll
      for (int j = 0; j < 8; ++j) {
        const int rl = mi * 16 + j + ((lane & 16) ? 8 : 0);
        const int rg = i0 + rl;
        const float v = Sacc[ni][j] * ATTN_SCALE;
        const bool ok = (cg <= rg) && (rg - cg < WIN);
        Ss[rl * 66 + cl] = ok ? v : -1e30f;
      }
    }
    __syncthreads();

    // ---- online softmax, one thread per row ----
    if (tid < 64) {
      const float* row = Ss + tid * 66;
      float mx = mrow[tid];
      for (int c = 0; c < 64; ++c) mx = fmaxf(mx, row[c]);
      const float alpha = __expf(mrow[tid] - mx);
      __bf16* prow = Ps + tid * 68;
      float psum = 0.f;
      for (int c = 0; c < 64; ++c) {
        const float sv = row[c];
        const float p = (sv > -1e29f) ? __expf(sv - mx) : 0.f;
        psum += p;
        prow[c] = (__bf16)p;
      }
      mrow[tid] = mx;
      lrow[tid] = lrow[tid] * alpha + psum;
      arow[tid] = alpha;
    }
    __syncthreads();

    // ---- rescale O, then O += P * V ----
#pragma unroll
    for (int h = 0; h < 4; ++h)
#pragma unroll
      for (int j = 0; j < 8; ++j) {
        const int r = mi * 16 + j + ((lane & 16) ? 8 : 0);
        Oacc[h][j] *= arow[r];
      }
#pragma unroll
    for (int ks = 0; ks < 2; ++ks) {
      const v16bf a = load_frag_a(Ps + mi * 16 * 68 + ks * 32, 68, lane);
#pragma unroll
      for (int h = 0; h < 4; ++h) {
        const v16bf b = load_frag_b(Vt + (hbase + h * 16) * 68 + ks * 32, 68, lane);
        Oacc[h] = wmma_bf16(a, b, Oacc[h]);
      }
    }
    __syncthreads();
  }

  // ---- epilogue: O / l -> enc (bf16) ----
#pragma unroll
  for (int h = 0; h < 4; ++h) {
    const int cc = hbase + h * 16 + (lane & 15);
#pragma unroll
    for (int j = 0; j < 8; ++j) {
      const int r = mi * 16 + j + ((lane & 16) ? 8 : 0);
      const float l = lrow[r];
      const float inv = (l > 0.f) ? 1.f / l : 0.f;
      encb[(long long)(i0 + r) * (NH * HH) + n * HH + cc] =
          (__bf16)(Oacc[h][j] * inv);
    }
  }
}

// ---------------------------------------------------------------------------
// host side
// ---------------------------------------------------------------------------
extern "C" void kernel_launch(void* const* d_in, const int* in_sizes, int n_in,
                              void* d_out, int out_size, void* d_ws, size_t ws_size,
                              hipStream_t stream) {
  (void)in_sizes; (void)n_in; (void)out_size; (void)ws_size;
  const float* x   = (const float*)d_in[0];  // [T, D]
  const float* qw  = (const float*)d_in[1];  // [N, D, H]
  const float* kvw = (const float*)d_in[2];  // [2, K, D, H]
  const float* ow  = (const float*)d_in[3];  // [N, H, D]
  const float* qns = (const float*)d_in[4];  // [H]
  const float* kns = (const float*)d_in[5];  // [H]
  float* out = (float*)d_out;                // [T, D] f32

  // workspace carve-up (bf16 buffers), ~92 MB total
  char* p = (char*)d_ws;
  __bf16* xb   = (__bf16*)p; p += (size_t)TT * DD * 2;
  __bf16* qwb  = (__bf16*)p; p += (size_t)NH * DD * HH * 2;
  __bf16* kvwb = (__bf16*)p; p += (size_t)2 * KVH * DD * HH * 2;
  __bf16* owb  = (__bf16*)p; p += (size_t)NH * HH * DD * 2;
  __bf16* qb   = (__bf16*)p; p += (size_t)TT * NH * HH * 2;
  __bf16* kb   = (__bf16*)p; p += (size_t)TT * KVH * HH * 2;
  __bf16* vb   = (__bf16*)p; p += (size_t)TT * KVH * HH * 2;
  __bf16* encb = (__bf16*)p; p += (size_t)TT * NH * HH * 2;

  // 1) f32 -> bf16 conversions
  {
    const long long nx = (long long)TT * DD;
    const long long nq = (long long)NH * DD * HH;
    const long long nk = (long long)2 * KVH * DD * HH;
    const long long no = (long long)NH * HH * DD;
    cvt_f32_bf16<<<(unsigned)((nx + 255) / 256), 256, 0, stream>>>(x, xb, nx);
    cvt_f32_bf16<<<(unsigned)((nq + 255) / 256), 256, 0, stream>>>(qw, qwb, nq);
    cvt_f32_bf16<<<(unsigned)((nk + 255) / 256), 256, 0, stream>>>(kvw, kvwb, nk);
    cvt_f32_bf16<<<(unsigned)((no + 255) / 256), 256, 0, stream>>>(ow, owb, no);
  }

  // 2) projections, q/k with fused RMSNorm+RoPE epilogue
  gemm64x128<2><<<dim3(TT / 64, NH), 256, 0, stream>>>(
      xb, DD, qwb, (long long)DD * HH, HH, DD,
      (void*)qb, (long long)HH, (long long)(NH * HH), qns);
  gemm64x128<2><<<dim3(TT / 64, KVH), 256, 0, stream>>>(
      xb, DD, kvwb, (long long)DD * HH, HH, DD,
      (void*)kb, (long long)HH, (long long)(KVH * HH), kns);
  gemm64x128<1><<<dim3(TT / 64, KVH), 256, 0, stream>>>(
      xb, DD, kvwb + (size_t)KVH * DD * HH, (long long)DD * HH, HH, DD,
      (void*)vb, (long long)HH, (long long)(KVH * HH), nullptr);

  // 3) flash attention -> encb [T, N, H]
  attn_kernel<<<dim3(TT / 64, NH), 256, 0, stream>>>(qb, kb, vb, encb);

  // 4) output projection: [T, N*H] x [N*H, D] -> d_out f32
  gemm64x128<0><<<dim3(TT / 64, DD / 128), 256, 0, stream>>>(
      encb, NH * HH, owb, (long long)128, DD, NH * HH,
      (void*)out, (long long)128, (long long)DD, nullptr);
}
